// RBFKernelProvider_16123307229754
// MI455X (gfx1250) — matrix operand order repaired and checked by reading, not compiled, not run on hardware
//
#include <hip/hip_runtime.h>
#include <hip/hip_bf16.h>
#include <math.h>

// RBF kernel matrix: out[n,m] = amp^2 * exp(-0.5 * max(|xs_n|^2 + |ys_m|^2 - 2 xs_n.ys_m, 0))
// GEMM done with v_wmma_f32_16x16x32_bf16 using a bf16 hi/lo split (3 products)
// to recover fp32-class accuracy. Operands are pre-swizzled into the exact
// per-lane WMMA fragment layouts (CDNA5 ISA 7.12.2) so fragment loads are
// coalesced b128 loads straight from global memory (L2-resident, 32 MB total).

typedef __attribute__((ext_vector_type(16))) __bf16 v16bf;
typedef __attribute__((ext_vector_type(8)))  float  v8f;

#define N_ROWS   8192
#define M_COLS   8192
#define DDIM     512
#define KT_TILES (DDIM / 32)   // 16 K-tiles of 32

__device__ __forceinline__ float softplus_f(float x) { return log1pf(expf(x)); }

// ---------------------------------------------------------------------------
// Kernel 0: parameter transforms (softplus + TINY), amp^2 and 1/length_scale
// ---------------------------------------------------------------------------
__global__ void rbf_params_kernel(const float* __restrict__ amp_raw,
                                  const float* __restrict__ ls_raw,
                                  float* __restrict__ amp2,
                                  float* __restrict__ inv_ls) {
  const float TINY = 1.17549435e-38f;  // smallest normal fp32 (jnp.finfo.tiny)
  int t = blockIdx.x * blockDim.x + threadIdx.x;
  if (t < DDIM) {
    float l = softplus_f(ls_raw[t]) + TINY;
    inv_ls[t] = 1.0f / l;
  }
  if (t == 0) {
    float a = softplus_f(amp_raw[0]) + TINY;
    amp2[0] = a * a;
  }
}

// ---------------------------------------------------------------------------
// Kernel 1: scale rows by inv_ls, compute fp32 row norms, split into bf16
// hi/lo, and scatter into WMMA fragment-major layout.
//   A layout (16-bit A 16x32): lane = m + 16*((k>>3)&1), slot = (k&7)+8*(k>>4)
//   B layout (16-bit B 32x16): lane = m + 16*((k>>3)&1), slot = (k&7)+8*(k>>4)
// One wave per row; in iteration t the lane's element has k == lane, so the
// (lane_dst, slot) mapping is a per-lane constant.
// ---------------------------------------------------------------------------
__global__ __launch_bounds__(256) void rbf_pack_kernel(
    const float* __restrict__ src, const float* __restrict__ inv_ls,
    __bf16* __restrict__ hi, __bf16* __restrict__ lo,
    float* __restrict__ sq, int b_layout) {
  const int lane = threadIdx.x & 31;
  const int row  = blockIdx.x * 8 + (threadIdx.x >> 5);
  const int m    = row & 15;
  const int tm   = row >> 4;

  int lane_dst, slot;
  if (b_layout) {
    lane_dst = m + 16 * ((lane >> 3) & 1);
    slot     = (lane & 7) + 8 * (lane >> 4);
  } else {
    lane_dst = m + 16 * ((lane >> 3) & 1);
    slot     = (lane & 7) + 8 * (lane >> 4);
  }

  float norm = 0.0f;
#pragma unroll
  for (int t = 0; t < KT_TILES; ++t) {
    int d = t * 32 + lane;
    float v = src[row * DDIM + d] * inv_ls[d];
    norm = fmaf(v, v, norm);
    __bf16 h = (__bf16)v;
    __bf16 l = (__bf16)(v - (float)h);
    int idx = ((tm * KT_TILES + t) << 9) + (lane_dst << 4) + slot;
    hi[idx] = h;
    lo[idx] = l;
  }
  // wave32 reduction for the fp32 row norm
  for (int off = 16; off > 0; off >>= 1)
    norm += __shfl_down(norm, off, 32);
  if (lane == 0) sq[row] = norm;
}

// ---------------------------------------------------------------------------
// Kernel 2: WMMA GEMM + RBF epilogue.
// Block tile 128x128, 8 waves arranged 2x4; each wave: 64x32 tile = 4x2
// 16x16 f32 accumulators. K loop: 16 steps of 32; 3 bf16 WMMAs per subtile
// per step (hi*hi + hi*lo + lo*hi).
// ---------------------------------------------------------------------------
__global__ __launch_bounds__(256) void rbf_wmma_kernel(
    const __bf16* __restrict__ Ahi, const __bf16* __restrict__ Alo,
    const __bf16* __restrict__ Bhi, const __bf16* __restrict__ Blo,
    const float* __restrict__ xsq, const float* __restrict__ ysq,
    const float* __restrict__ amp2p, float* __restrict__ out) {
  const int lane = threadIdx.x & 31;
  const int w    = threadIdx.x >> 5;   // 0..7
  const int wr   = w >> 2;             // 0..1  (row group of waves)
  const int wc   = w & 3;              // 0..3  (col group of waves)
  const int row0 = blockIdx.y * 128 + wr * 64;
  const int col0 = blockIdx.x * 128 + wc * 32;
  const int tm0  = row0 >> 4;          // first 16-row A tile
  const int tn0  = col0 >> 4;          // first 16-col B tile

  v8f acc[4][2] = {};

  for (int kt = 0; kt < KT_TILES; ++kt) {
    v16bf ah[4], al[4], bh[2], bl[2];
#pragma unroll
    for (int i = 0; i < 4; ++i) {
      int base = (((tm0 + i) * KT_TILES + kt) << 9) + (lane << 4);
      ah[i] = *(const v16bf*)(Ahi + base);
      al[i] = *(const v16bf*)(Alo + base);
    }
#pragma unroll
    for (int j = 0; j < 2; ++j) {
      int base = (((tn0 + j) * KT_TILES + kt) << 9) + (lane << 4);
      bh[j] = *(const v16bf*)(Bhi + base);
      bl[j] = *(const v16bf*)(Blo + base);
    }
#pragma unroll
    for (int i = 0; i < 4; ++i) {
#pragma unroll
      for (int j = 0; j < 2; ++j) {
        acc[i][j] = __builtin_amdgcn_wmma_f32_16x16x32_bf16(
            false, ah[i], false, bh[j], (short)0, acc[i][j], false, false);
        acc[i][j] = __builtin_amdgcn_wmma_f32_16x16x32_bf16(
            false, ah[i], false, bl[j], (short)0, acc[i][j], false, false);
        acc[i][j] = __builtin_amdgcn_wmma_f32_16x16x32_bf16(
            false, al[i], false, bh[j], (short)0, acc[i][j], false, false);
      }
    }
  }

  // Epilogue: C/D layout -> VGPR v, lane L: row = v + 8*(L>=16), col = L&15
  const float a2   = amp2p[0];
  const int   cl   = lane & 15;
  const int   half = lane >> 4;
  const float NEG_HALF_LOG2E = -0.72134752044448170367f;  // -0.5*log2(e)
#pragma unroll
  for (int i = 0; i < 4; ++i) {
#pragma unroll
    for (int j = 0; j < 2; ++j) {
      int col  = col0 + j * 16 + cl;
      float yc = ysq[col];
#pragma unroll
      for (int v = 0; v < 8; ++v) {
        int r = row0 + i * 16 + v + 8 * half;
        float sq = fmaxf(xsq[r] + yc - 2.0f * acc[i][j][v], 0.0f);
        out[(size_t)r * M_COLS + col] = a2 * exp2f(sq * NEG_HALF_LOG2E);
      }
    }
  }
}

// ---------------------------------------------------------------------------
// Host-side launch
// ---------------------------------------------------------------------------
extern "C" void kernel_launch(void* const* d_in, const int* in_sizes, int n_in,
                              void* d_out, int out_size, void* d_ws, size_t ws_size,
                              hipStream_t stream) {
  (void)in_sizes; (void)n_in; (void)out_size; (void)ws_size;
  const float* x  = (const float*)d_in[0];  // (N, D)
  const float* x2 = (const float*)d_in[1];  // (M, D)
  const float* ar = (const float*)d_in[2];  // amplitude_raw (1,)
  const float* lr = (const float*)d_in[3];  // length_scale_raw (D,)
  float* out = (float*)d_out;

  char* ws = (char*)d_ws;
  float*  amp2   = (float*)(ws + 0);
  float*  inv_ls = (float*)(ws + 256);
  float*  xsq    = (float*)(ws + 4096);                 // 8192 floats
  float*  ysq    = (float*)(ws + 4096 + 32768);         // 8192 floats
  __bf16* Ahi    = (__bf16*)(ws + 131072);              // N*D bf16 = 8 MB each
  __bf16* Alo    = Ahi + (size_t)N_ROWS * DDIM;
  __bf16* Bhi    = Alo + (size_t)N_ROWS * DDIM;
  __bf16* Blo    = Bhi + (size_t)M_COLS * DDIM;

  hipLaunchKernelGGL(rbf_params_kernel, dim3(2), dim3(256), 0, stream,
                     ar, lr, amp2, inv_ls);
  hipLaunchKernelGGL(rbf_pack_kernel, dim3(N_ROWS / 8), dim3(256), 0, stream,
                     x, inv_ls, Ahi, Alo, xsq, 0);
  hipLaunchKernelGGL(rbf_pack_kernel, dim3(M_COLS / 8), dim3(256), 0, stream,
                     x2, inv_ls, Bhi, Blo, ysq, 1);
  hipLaunchKernelGGL(rbf_wmma_kernel, dim3(M_COLS / 128, N_ROWS / 128),
                     dim3(256), 0, stream,
                     Ahi, Alo, Bhi, Blo, xsq, ysq, amp2, out);
}
